// LocalAttentionHead_20907900797368
// MI455X (gfx1250) — matrix operand order repaired and checked by reading, not compiled, not run on hardware
//
#include <hip/hip_runtime.h>

typedef __attribute__((ext_vector_type(16))) __bf16 v16bf;
typedef __attribute__((ext_vector_type(8)))  float  v8f;
typedef __attribute__((ext_vector_type(4)))  unsigned int u32x4;
typedef __attribute__((ext_vector_type(4)))  int i32x4;

#define B_    256
#define NTOK  196
#define DIM_  768
#define H_    12
#define HD_   64
#define NPQ   208   // queries padded to 13*16
#define NPK   224   // keys padded to 14*16 (7 K-steps of 32)
#define QT    (NPQ/16)
#define KT    (NPK/16)

// LDS row strides (bf16 elems), chosen so byte stride is a multiple of 16
// (b128-aligned) and lane bank-starts are all distinct mod 64 banks.
#define GASTR 40    // gemm tiles: 80B rows
#define QSTR  72    // Q/K tiles: 144B rows
#define VSTR  232   // V^T / P tiles: 464B rows

#if defined(__AMDGCN__) && __has_builtin(__builtin_amdgcn_global_load_async_to_lds_b128)
#define ASYNC_LDS 1
#else
#define ASYNC_LDS 0
#endif

__device__ __forceinline__ v8f wmma_bf16(v16bf a, v16bf b, v8f c) {
  return __builtin_amdgcn_wmma_f32_16x16x32_bf16(false, a, false, b, (short)0, c,
                                                 false, false);
}

// A-fragment element->K mapping for 16-bit 16x32 A operand (wave32):
// lanes 0-15 hold K 0-15; lanes 16-31 hold K 16-31 (contiguous, matches B).
__device__ __forceinline__ int akmap(int e, int lh) {
  return lh * 16 + e;
}

__device__ __forceinline__ void async_wait_all() {
#if ASYNC_LDS
#if __has_builtin(__builtin_amdgcn_s_wait_asynccnt)
  __builtin_amdgcn_s_wait_asynccnt(0);
#else
  asm volatile("s_wait_asynccnt 0x0" ::: "memory");
#endif
#endif
}

// ---------------- fp32 -> bf16 pre-convert (n multiple of 4) -----------------
__global__ __launch_bounds__(256) void cvt_bf16_kernel(
    const float* __restrict__ src, __bf16* __restrict__ dst, int n) {
  int i = (blockIdx.x * 256 + threadIdx.x) * 4;
  if (i >= n) return;
  float4 v = *(const float4*)(src + i);
  dst[i + 0] = (__bf16)v.x;
  dst[i + 1] = (__bf16)v.y;
  dst[i + 2] = (__bf16)v.z;
  dst[i + 3] = (__bf16)v.w;
}

// ---------------- bias plane: mask + rpb gather, padded [H][NPQ][NPK] --------
__global__ __launch_bounds__(256) void bias_prep_kernel(
    const float* __restrict__ rpb, const unsigned char* __restrict__ mask,
    const int* __restrict__ rel_idx, float* __restrict__ biasm) {
  int idx = blockIdx.x * 256 + threadIdx.x;
  if (idx >= H_ * NPQ * NPK) return;
  int k = idx % NPK;
  int q = (idx / NPK) % NPQ;
  int h = idx / (NPK * NPQ);
  float v = -1.0e30f;
  if (q < NTOK && k < NTOK) {
    int qi = q * NTOK + k;
    if (!mask[qi]) v = rpb[rel_idx[qi] * H_ + h];
  }
  biasm[idx] = v;
}

// copy a 128x32 bf16 panel (row stride K) into LDS (row stride GASTR),
// 16B per lane, 2 chunks per thread; async direct-to-LDS when available.
__device__ __forceinline__ void stage_tile(const __bf16* __restrict__ g,
                                           __bf16* lds, int K, int tid) {
#pragma unroll
  for (int c = tid; c < 512; c += 256) {
    int row = c >> 2, sect = c & 3;           // 4 chunks of 8 bf16 per row
    const __bf16* src = g + (size_t)row * K + sect * 8;
    __bf16* dst = lds + row * GASTR + sect * 8;
#if ASYNC_LDS
    __builtin_amdgcn_global_load_async_to_lds_b128(
        (__attribute__((address_space(1))) i32x4*)src,
        (__attribute__((address_space(3))) i32x4*)dst, 0, 0);
#else
    *(u32x4*)dst = *(const u32x4*)src;
#endif
  }
}

// ---------------- generic GEMM: C[m,o] = sum_k A[m,k] * W[o,k] ---------------
// A, W both bf16. EPI==0: scatter bf16 into qkv layout [3,B,H,NTOK,HD] with
// q scaled by 0.125.  EPI==1: fp32 out[m*DIM_+o] = acc + bvec[o].
template <int EPI>
__global__ __launch_bounds__(256) void gemm_kernel(
    const __bf16* __restrict__ A, const __bf16* __restrict__ W,
    const float* __restrict__ bvec, void* __restrict__ outp, int K) {
  __shared__ __bf16 As[2][128 * GASTR];
  __shared__ __bf16 Bs[2][128 * GASTR];

  int tid  = threadIdx.x;
  int lane = tid & 31, wave = tid >> 5;
  int wm = wave >> 1, wn = wave & 1;     // 4x2 wave grid over 128x128 tile
  int lh = lane >> 4, ll = lane & 15;
  int m0 = blockIdx.x * 128, n0 = blockIdx.y * 128;

  v8f zero = {0.f, 0.f, 0.f, 0.f, 0.f, 0.f, 0.f, 0.f};
  v8f acc[2][4];
#pragma unroll
  for (int i = 0; i < 2; ++i)
#pragma unroll
    for (int j = 0; j < 4; ++j) acc[i][j] = zero;

  const __bf16* Ab = A + (size_t)m0 * K;
  const __bf16* Wb = W + (size_t)n0 * K;

  stage_tile(Ab, As[0], K, tid);
  stage_tile(Wb, Bs[0], K, tid);
  async_wait_all();
  __syncthreads();

  int cur = 0;
  for (int k0 = 0; k0 < K; k0 += 32) {
    if (k0 + 32 < K) {                      // prefetch next panel
      stage_tile(Ab + k0 + 32, As[cur ^ 1], K, tid);
      stage_tile(Wb + k0 + 32, Bs[cur ^ 1], K, tid);
    }

    v16bf af[2], bfm[4];
#pragma unroll
    for (int i = 0; i < 2; ++i) {
      const __bf16* ap = As[cur] + (wm * 32 + i * 16 + ll) * GASTR;
#pragma unroll
      for (int e = 0; e < 16; ++e) af[i][e] = ap[akmap(e, lh)];
    }
#pragma unroll
    for (int j = 0; j < 4; ++j) {
      const __bf16* bp = Bs[cur] + (wn * 64 + j * 16 + ll) * GASTR + lh * 16;
#pragma unroll
      for (int e = 0; e < 16; ++e) bfm[j][e] = bp[e];
    }
#pragma unroll
    for (int i = 0; i < 2; ++i)
#pragma unroll
      for (int j = 0; j < 4; ++j) acc[i][j] = wmma_bf16(af[i], bfm[j], acc[i][j]);

    async_wait_all();
    __syncthreads();
    cur ^= 1;
  }

#pragma unroll
  for (int i = 0; i < 2; ++i) {
#pragma unroll
    for (int j = 0; j < 4; ++j) {
#pragma unroll
      for (int r = 0; r < 8; ++r) {
        int gm = m0 + wm * 32 + i * 16 + lh * 8 + r;
        int gn = n0 + wn * 64 + j * 16 + ll;
        float v = acc[i][j][r];
        if (EPI == 0) {
          int bb  = gm / NTOK,  t   = gm - bb * NTOK;
          int sdx = gn / DIM_,  rem = gn - sdx * DIM_;
          int hh  = rem >> 6,   hd  = rem & 63;
          if (sdx == 0) v *= 0.125f;   // HD^-0.5, exact in bf16
          ((__bf16*)outp)[((((size_t)sdx * B_ + bb) * H_ + hh) * NTOK + t) * HD_ + hd] =
              (__bf16)v;
        } else {
          ((float*)outp)[(size_t)gm * DIM_ + gn] = v + bvec[gn];
        }
      }
    }
  }
}

// ---------------- fused attention: one workgroup per (b,h) -------------------
__global__ __launch_bounds__(256) void attn_kernel(
    const __bf16* __restrict__ qkv, const float* __restrict__ biasm,
    __bf16* __restrict__ ao) {
  __shared__ __bf16 Qs[NPQ * QSTR];       // [query][hd], padded rows
  __shared__ __bf16 Ks[NPK * QSTR];       // [key][hd], padded rows
  __shared__ __bf16 Vt[HD_ * VSTR];       // [hd][key] (transposed), padded rows
  __shared__ __bf16 Ps[8 * 16 * VSTR];    // per-wave probability rows

  int tid  = threadIdx.x;
  int lane = tid & 31, wave = tid >> 5;
  int lh = lane >> 4, ll = lane & 15;
  int b = blockIdx.x / H_, h = blockIdx.x % H_;

  size_t base  = ((size_t)b * H_ + h) * (NTOK * HD_);
  size_t plane = (size_t)B_ * H_ * NTOK * HD_;
  const __bf16* qg = qkv + base;
  const __bf16* kg = qkv + plane + base;
  const __bf16* vg = qkv + 2 * plane + base;

  for (int idx = tid; idx < NPQ * HD_; idx += 256) {
    int row = idx >> 6, hd = idx & 63;
    Qs[row * QSTR + hd] = (row < NTOK) ? qg[idx] : (__bf16)0.0f;
  }
  for (int idx = tid; idx < NPK * HD_; idx += 256) {
    int row = idx >> 6, hd = idx & 63;
    Ks[row * QSTR + hd] = (row < NTOK) ? kg[idx] : (__bf16)0.0f;
  }
  for (int idx = tid; idx < NPK * HD_; idx += 256) {
    int key = idx >> 6, hd = idx & 63;
    Vt[hd * VSTR + key] = (key < NTOK) ? vg[idx] : (__bf16)0.0f;
  }
  __syncthreads();

  const float* bp = biasm + (size_t)h * NPQ * NPK;
  __bf16* pw = Ps + wave * (16 * VSTR);
  v8f zero = {0.f, 0.f, 0.f, 0.f, 0.f, 0.f, 0.f, 0.f};

  for (int rt = wave; rt < QT; rt += 8) {
    int q0 = rt * 16;
    v8f s[KT];
#pragma unroll
    for (int t = 0; t < KT; ++t) s[t] = zero;

    // S = Q K^T  (HD=64 -> 2 K-steps of 32)
#pragma unroll
    for (int ks = 0; ks < 2; ++ks) {
      v16bf a;
      const __bf16* qp = Qs + (q0 + ll) * QSTR + ks * 32;
#pragma unroll
      for (int e = 0; e < 16; ++e) a[e] = qp[akmap(e, lh)];
#pragma unroll
      for (int t = 0; t < KT; ++t) {
        v16bf bb;
        const __bf16* kp = Ks + (t * 16 + ll) * QSTR + ks * 32 + lh * 16;
#pragma unroll
        for (int e = 0; e < 16; ++e) bb[e] = kp[e];
        s[t] = wmma_bf16(a, bb, s[t]);
      }
    }

    // mask + relative-position bias (pre-folded, padded cols are -1e30)
#pragma unroll
    for (int t = 0; t < KT; ++t)
#pragma unroll
      for (int r = 0; r < 8; ++r)
        s[t][r] += bp[(q0 + lh * 8 + r) * NPK + t * 16 + ll];

    // softmax per row: each row lives in 16 lanes of one half-wave
#pragma unroll
    for (int r = 0; r < 8; ++r) {
      float m = -3.0e38f;
#pragma unroll
      for (int t = 0; t < KT; ++t) m = fmaxf(m, s[t][r]);
#pragma unroll
      for (int off = 1; off < 16; off <<= 1) m = fmaxf(m, __shfl_xor(m, off, 32));
      float sum = 0.f;
#pragma unroll
      for (int t = 0; t < KT; ++t) {
        float e = __expf(s[t][r] - m);
        s[t][r] = e;
        sum += e;
      }
#pragma unroll
      for (int off = 1; off < 16; off <<= 1) sum += __shfl_xor(sum, off, 32);
      float inv = 1.f / sum;
#pragma unroll
      for (int t = 0; t < KT; ++t) s[t][r] *= inv;
    }

    // stash P (bf16) in this wave's private LDS strip, re-read as A fragments
#pragma unroll
    for (int t = 0; t < KT; ++t)
#pragma unroll
      for (int r = 0; r < 8; ++r)
        pw[(lh * 8 + r) * VSTR + t * 16 + ll] = (__bf16)s[t][r];
    asm volatile("s_wait_dscnt 0" ::: "memory");  // same-wave LDS RAW

    // O = P V  (224 keys -> 7 K-steps of 32)
    v8f o[4];
#pragma unroll
    for (int j = 0; j < 4; ++j) o[j] = zero;
#pragma unroll
    for (int ks = 0; ks < KT / 2; ++ks) {
      v16bf a;
      const __bf16* pp = pw + ll * VSTR + ks * 32;
#pragma unroll
      for (int e = 0; e < 16; ++e) a[e] = pp[akmap(e, lh)];
#pragma unroll
      for (int j = 0; j < 4; ++j) {
        v16bf bb;
        const __bf16* vp = Vt + (j * 16 + ll) * VSTR + ks * 32 + lh * 16;
#pragma unroll
        for (int e = 0; e < 16; ++e) bb[e] = vp[e];
        o[j] = wmma_bf16(a, bb, o[j]);
      }
    }

#pragma unroll
    for (int j = 0; j < 4; ++j)
#pragma unroll
      for (int r = 0; r < 8; ++r) {
        int q = q0 + lh * 8 + r;
        if (q < NTOK)
          ao[((size_t)b * NTOK + q) * DIM_ + h * HD_ + j * 16 + ll] = (__bf16)o[j][r];
      }
  }
}

extern "C" void kernel_launch(void* const* d_in, const int* in_sizes, int n_in,
                              void* d_out, int out_size, void* d_ws, size_t ws_size,
                              hipStream_t stream) {
  (void)in_sizes; (void)n_in; (void)out_size; (void)ws_size;
  const float* x      = (const float*)d_in[0];
  const float* w_qkv  = (const float*)d_in[1];
  const float* rpb    = (const float*)d_in[2];
  const float* proj_w = (const float*)d_in[3];
  const float* proj_b = (const float*)d_in[4];
  const unsigned char* mask = (const unsigned char*)d_in[5];
  const int* rel_idx  = (const int*)d_in[6];
  float* out = (float*)d_out;

  // workspace layout (elements):
  //   qkv bf16 [3,B,H,NTOK,HD] | attn-out bf16 [B,NTOK,DIM] | bias f32 |
  //   x bf16 | w_qkv bf16 | proj_w bf16
  __bf16* qkv_ws  = (__bf16*)d_ws;
  size_t  qkv_n   = (size_t)3 * B_ * H_ * NTOK * HD_;
  __bf16* attn_ws = qkv_ws + qkv_n;
  size_t  ao_n    = (size_t)B_ * NTOK * DIM_;
  float*  biasm   = (float*)(attn_ws + ao_n);
  size_t  bias_n  = (size_t)H_ * NPQ * NPK;
  __bf16* xbf     = (__bf16*)(biasm + bias_n);
  size_t  x_n     = (size_t)B_ * NTOK * DIM_;
  __bf16* wqkvbf  = xbf + x_n;
  size_t  wq_n    = (size_t)3 * DIM_ * DIM_;
  __bf16* projwbf = wqkvbf + wq_n;
  size_t  pw_n    = (size_t)DIM_ * DIM_;

  cvt_bf16_kernel<<<(int)(x_n / 1024), 256, 0, stream>>>(x, xbf, (int)x_n);
  cvt_bf16_kernel<<<(int)(wq_n / 1024), 256, 0, stream>>>(w_qkv, wqkvbf, (int)wq_n);
  cvt_bf16_kernel<<<(int)(pw_n / 1024), 256, 0, stream>>>(proj_w, projwbf, (int)pw_n);

  int btotal = (int)bias_n;
  bias_prep_kernel<<<(btotal + 255) / 256, 256, 0, stream>>>(rpb, mask, rel_idx, biasm);

  dim3 g1((B_ * NTOK) / 128, (3 * DIM_) / 128);   // 392 x 18
  gemm_kernel<0><<<g1, 256, 0, stream>>>(xbf, wqkvbf, nullptr, (void*)qkv_ws, DIM_);

  attn_kernel<<<B_ * H_, 256, 0, stream>>>(qkv_ws, biasm, attn_ws);

  dim3 g3((B_ * NTOK) / 128, DIM_ / 128);         // 392 x 6
  gemm_kernel<1><<<g3, 256, 0, stream>>>(attn_ws, projwbf, proj_b, (void*)out, DIM_);
}
